// ChannelExtensionGlobalAttentionFeatureEncoder_82437602280033
// MI455X (gfx1250) — matrix operand order repaired and checked by reading, not compiled, not run on hardware
//
#include <hip/hip_runtime.h>

typedef __attribute__((ext_vector_type(16))) __bf16 v16bf;
typedef __attribute__((ext_vector_type(8)))  __bf16 v8bf;
typedef __attribute__((ext_vector_type(8)))  float  v8f;
typedef int v4i __attribute__((vector_size(16)));

#define GLOBAL_AS __attribute__((address_space(1)))
#define LDS_AS    __attribute__((address_space(3)))

// gfx1250 async global->LDS copy path (probe-guarded; falls back to ds stores)
#if defined(__has_builtin)
#if __has_builtin(__builtin_amdgcn_global_load_async_to_lds_b128) && \
    __has_builtin(__builtin_amdgcn_s_wait_asynccnt)
#define HAVE_ASYNC_LDS 1
#endif
#endif
#ifndef HAVE_ASYNC_LDS
#define HAVE_ASYNC_LDS 0
#endif

// ---------------- problem constants ----------------
constexpr int BN = 2, CC = 128, HN = 96, WN = 192;
constexpr int NTOK = BN * HN * WN;        // 36864 tokens
constexpr int NS = 4, WH = HN / NS, WW = WN / NS;   // 24, 48
constexpr int WTOK = WH * WW;             // 1152 tokens / window
constexpr int NWIN = BN * NS * NS;        // 32 windows
constexpr int SH = WH / 2, SW = WW / 2;   // 12, 24
constexpr float TWO_PI = 6.28318530717958647692f;

// gemm flags (epilogue behavior; data layout is templated)
#define GF_A_F32 1
#define GF_B_NT  2
#define GF_GELU  4
#define GF_RES   8
#define GF_MASK  16

__device__ __forceinline__ float gelu_tanh(float x) {
    float x3 = x * x * x;
    return 0.5f * x * (1.f + tanhf(0.79788456080286535588f * (x + 0.044715f * x3)));
}

// ---------------- generic batched WMMA GEMM ----------------
// C[M,N] = act( scale*(A[M,K] @ op(B)) + bias ) (+mask) (+res), fp32 accum.
// A: bf16 row-major (lda) or f32 row-major (AF32, converted while staging LDS)
// B: NN -> bf16 [K x ldb];  NT -> bf16 [N x ldb] (row-major B^T, ldb = K stride)
// Workgroup: 256 threads = 8 waves; macro-tile 128(M) x 128(N); wave tile 32x64;
// K-step 32 -> 8 v_wmma_f32_16x16x32_bf16 per wave per step.
template <bool AF32, bool BNT>
__global__ __launch_bounds__(256)
void gemm_kernel(const void* __restrict__ Ap, const __bf16* __restrict__ Bp,
                 const float* __restrict__ bias,
                 float* __restrict__ Cf, __bf16* __restrict__ Ch,
                 const float* __restrict__ res,
                 int M, int N, int K, int lda, int ldb, int ldc,
                 long long sA, long long sB, long long sC,
                 float scale, int flags)
{
    const int t  = threadIdx.x;
    const int m0 = blockIdx.y * 128;
    const int n0 = blockIdx.x * 128;
    const int bz = blockIdx.z;

    const int wv = t >> 5, lane = t & 31;
    const int lr = lane & 15, hi = lane >> 4;
    const int wm = wv >> 1, wn = wv & 1;    // 4(M) x 2(N) wave grid

    v8f acc[2][4];
    #pragma unroll
    for (int i = 0; i < 2; ++i)
        #pragma unroll
        for (int j = 0; j < 4; ++j)
            #pragma unroll
            for (int r = 0; r < 8; ++r) ((float*)&acc[i][j])[r] = 0.f;

    const __bf16* Ab = AF32 ? nullptr : ((const __bf16*)Ap + bz * sA);
    const float*  Af = AF32 ? ((const float*)Ap + bz * sA) : nullptr;
    const __bf16* Bb = Bp + bz * sB;

    // fragment gather + 8 WMMAs for one K-step (LDS stride = 40 halves, 16B aligned)
    auto compute = [&](const __bf16* Asp, const __bf16* Bsp) {
        v16bf afr[2], bfr[4];
        #pragma unroll
        for (int mi = 0; mi < 2; ++mi) {
            int row = wm * 32 + mi * 16 + lr;
            v8bf lo = *(const v8bf*)&Asp[row * 40 + hi * 8];
            v8bf hh = *(const v8bf*)&Asp[row * 40 + 16 + hi * 8];
            __bf16* f = (__bf16*)&afr[mi];
            *(v8bf*)f = lo; *(v8bf*)(f + 8) = hh;
        }
        #pragma unroll
        for (int ni = 0; ni < 4; ++ni) {
            int col = wn * 64 + ni * 16 + lr;
            v8bf lo = *(const v8bf*)&Bsp[col * 40 + hi * 8];
            v8bf hh = *(const v8bf*)&Bsp[col * 40 + 16 + hi * 8];
            __bf16* f = (__bf16*)&bfr[ni];
            *(v8bf*)f = lo; *(v8bf*)(f + 8) = hh;
        }
        #pragma unroll
        for (int mi = 0; mi < 2; ++mi)
            #pragma unroll
            for (int ni = 0; ni < 4; ++ni)
                acc[mi][ni] = __builtin_amdgcn_wmma_f32_16x16x32_bf16(
                    false, afr[mi], false, bfr[ni], (short)0, acc[mi][ni], false, false);
    };

#if HAVE_ASYNC_LDS
    if constexpr (!AF32 && BNT) {
        // double-buffered async global->LDS pipeline (gfx1250 ASYNCcnt path):
        // both A and B^T tiles are 128 rows x 32 halves, per-lane 16B contiguous.
        __shared__ __bf16 As[2][128 * 40];
        __shared__ __bf16 Bs[2][128 * 40];
        const int row = t >> 2, co = (t & 3) * 8;
        auto issue = [&](int k0, int buf) {
            #pragma unroll
            for (int rr = 0; rr < 2; ++rr) {
                __builtin_amdgcn_global_load_async_to_lds_b128(
                    (GLOBAL_AS v4i*)(Ab + (long long)(m0 + row + rr * 64) * lda + k0 + co),
                    (LDS_AS v4i*)&As[buf][(row + rr * 64) * 40 + co], 0, 0);
                __builtin_amdgcn_global_load_async_to_lds_b128(
                    (GLOBAL_AS v4i*)(Bb + (long long)(n0 + row + rr * 64) * ldb + k0 + co),
                    (LDS_AS v4i*)&Bs[buf][(row + rr * 64) * 40 + co], 0, 0);
            }
        };
        const int nk = K >> 5;
        issue(0, 0);
        for (int kt = 0; kt < nk; ++kt) {
            int cur = kt & 1;
            if (kt + 1 < nk) {
                issue((kt + 1) << 5, cur ^ 1);
                __builtin_amdgcn_s_wait_asynccnt(4);   // current tile's 4 copies complete
            } else {
                __builtin_amdgcn_s_wait_asynccnt(0);   // drain
            }
            __syncthreads();
            compute(&As[cur][0], &Bs[cur][0]);
            __syncthreads();   // all waves done with buf[cur] before it is refilled
        }
    } else
#endif
    {
        // single-buffer LDS with register-staged software pipeline
        __shared__ __bf16 As[128 * 40];
        __shared__ __bf16 Bs[128 * 40];
        float4 raf[4]; v8bf rah[2]; v8bf rb[2];

        auto loadA = [&](int k0) {
            if constexpr (AF32) {
                #pragma unroll
                for (int rr = 0; rr < 4; ++rr) {
                    int row = (t >> 3) + rr * 32, co = (t & 7) * 4;
                    raf[rr] = *(const float4*)(Af + (long long)(m0 + row) * lda + k0 + co);
                }
            } else {
                #pragma unroll
                for (int rr = 0; rr < 2; ++rr) {
                    int row = (t >> 2) + rr * 64, co = (t & 3) * 8;
                    rah[rr] = *(const v8bf*)(Ab + (long long)(m0 + row) * lda + k0 + co);
                }
                __builtin_prefetch(Ab + (long long)(m0 + (t >> 2)) * lda + k0 + 32, 0, 1);
            }
        };
        auto loadB = [&](int k0) {
            if constexpr (BNT) {
                #pragma unroll
                for (int rr = 0; rr < 2; ++rr) {
                    int row = (t >> 2) + rr * 64, co = (t & 3) * 8;
                    rb[rr] = *(const v8bf*)(Bb + (long long)(n0 + row) * ldb + k0 + co);
                }
            } else {
                #pragma unroll
                for (int rr = 0; rr < 2; ++rr) {
                    int kk = (t >> 4) + rr * 16, n8 = (t & 15) * 8;
                    rb[rr] = *(const v8bf*)(Bb + (long long)(k0 + kk) * ldb + n0 + n8);
                }
            }
        };

        loadA(0); loadB(0);
        for (int k0 = 0; k0 < K; k0 += 32) {
            __syncthreads();
            if constexpr (AF32) {
                #pragma unroll
                for (int rr = 0; rr < 4; ++rr) {
                    int row = (t >> 3) + rr * 32, co = (t & 7) * 4;
                    __bf16* d = &As[row * 40 + co];
                    d[0] = (__bf16)raf[rr].x; d[1] = (__bf16)raf[rr].y;
                    d[2] = (__bf16)raf[rr].z; d[3] = (__bf16)raf[rr].w;
                }
            } else {
                #pragma unroll
                for (int rr = 0; rr < 2; ++rr) {
                    int row = (t >> 2) + rr * 64, co = (t & 3) * 8;
                    *(v8bf*)&As[row * 40 + co] = rah[rr];
                }
            }
            if constexpr (BNT) {
                #pragma unroll
                for (int rr = 0; rr < 2; ++rr) {
                    int row = (t >> 2) + rr * 64, co = (t & 3) * 8;
                    *(v8bf*)&Bs[row * 40 + co] = rb[rr];
                }
            } else {
                #pragma unroll
                for (int rr = 0; rr < 2; ++rr) {
                    int kk = (t >> 4) + rr * 16, n8 = (t & 15) * 8;
                    #pragma unroll
                    for (int e = 0; e < 8; ++e) Bs[(n8 + e) * 40 + kk] = rb[rr][e];
                }
            }
            __syncthreads();
            if (k0 + 32 < K) { loadA(k0 + 32); loadB(k0 + 32); }  // overlap with WMMAs
            compute(As, Bs);
        }
    }

    // ---- epilogue ----
    #pragma unroll
    for (int mi = 0; mi < 2; ++mi) {
        #pragma unroll
        for (int ni = 0; ni < 4; ++ni) {
            int gcol = n0 + wn * 64 + ni * 16 + lr;
            if (gcol >= N) continue;
            float bb = bias ? bias[gcol] : 0.f;
            const float* av = (const float*)&acc[mi][ni];
            #pragma unroll
            for (int r = 0; r < 8; ++r) {
                int grow = m0 + wm * 32 + mi * 16 + hi * 8 + r;
                float v = av[r] * scale + bb;
                if (flags & GF_MASK) {   // swin region mask on scores
                    int win = bz & (NS * NS - 1);
                    int wi = win >> 2, wj = win & 3;
                    int hq = wi * WH + grow / WW, wq = wj * WW + grow % WW;
                    int hk = wi * WH + gcol / WW, xk = wj * WW + gcol % WW;
                    int rq = ((hq >= HN - WH) + (hq >= HN - SH)) * 3 + (wq >= WN - WW) + (wq >= WN - SW);
                    int rk = ((hk >= HN - WH) + (hk >= HN - SH)) * 3 + (xk >= WN - WW) + (xk >= WN - SW);
                    if (rq != rk) v += -1.0e9f;
                }
                if (flags & GF_GELU) v = gelu_tanh(v);
                long long ci = bz * sC + (long long)grow * ldc + gcol;
                if (flags & GF_RES) v += res[ci];
                if (Cf) Cf[ci] = v;
                if (Ch) Ch[ci] = (__bf16)v;
            }
        }
    }
}

// ---------------- LayerNorm: one wave per token (C=128, 4 per lane) ----------------
__global__ __launch_bounds__(256)
void ln_kernel(const float* __restrict__ x, const float* __restrict__ g,
               const float* __restrict__ b, __bf16* __restrict__ out, int ntok)
{
    int tok = blockIdx.x * 8 + (threadIdx.x >> 5);
    int lane = threadIdx.x & 31;
    if (tok >= ntok) return;
    float4 v = ((const float4*)(x + (long long)tok * 128))[lane];
    float s = v.x + v.y + v.z + v.w;
    float q = v.x * v.x + v.y * v.y + v.z * v.z + v.w * v.w;
    #pragma unroll
    for (int o = 16; o; o >>= 1) { s += __shfl_xor(s, o, 32); q += __shfl_xor(q, o, 32); }
    float mean = s * (1.f / 128.f);
    float var  = q * (1.f / 128.f) - mean * mean;
    float rstd = rsqrtf(var + 1e-5f);
    float4 gg = ((const float4*)g)[lane];
    float4 bv = ((const float4*)b)[lane];
    __bf16* o = out + (long long)tok * 128 + lane * 4;
    o[0] = (__bf16)((v.x - mean) * rstd * gg.x + bv.x);
    o[1] = (__bf16)((v.y - mean) * rstd * gg.y + bv.y);
    o[2] = (__bf16)((v.z - mean) * rstd * gg.z + bv.z);
    o[3] = (__bf16)((v.w - mean) * rstd * gg.w + bv.w);
}

// ---------------- weight f32 -> bf16 (with N padding) ----------------
__global__ void conv_w(const float* __restrict__ src, __bf16* __restrict__ dst,
                       int K, int N, int Npad)
{
    long long id = (long long)blockIdx.x * 256 + threadIdx.x;
    if (id >= (long long)K * Npad) return;
    int k = (int)(id / Npad), n = (int)(id % Npad);
    dst[id] = (n < N) ? (__bf16)src[(long long)k * N + n] : (__bf16)0.f;
}

// ---------------- layout kernels ----------------
__global__ void nchw_to_nhwc(const float* __restrict__ in, float* __restrict__ out)
{
    long long id = (long long)blockIdx.x * 256 + threadIdx.x;
    if (id >= (long long)BN * CC * HN * WN) return;
    int w = (int)(id % WN); long long r = id / WN;
    int h = (int)(r % HN); r /= HN;
    int c = (int)(r % CC); int b = (int)(r / CC);
    out[(((long long)(b * HN + h) * WN + w) << 7) + c] = in[id];
}
__global__ void nhwc_to_nchw(const float* __restrict__ in, float* __restrict__ out)
{
    long long id = (long long)blockIdx.x * 256 + threadIdx.x;
    if (id >= (long long)NTOK * 128) return;
    int c = (int)(id & 127); long long tok = id >> 7;
    int w = (int)(tok % WN); int h = (int)((tok / WN) % HN); int b = (int)(tok / (HN * WN));
    out[((long long)(b * CC + c) * HN + h) * WN + w] = in[id];
}

__global__ void fill_zero(float* __restrict__ p, long long n)
{
    long long id = (long long)blockIdx.x * 256 + threadIdx.x;
    if (id < n) p[id] = 0.f;
}

// ---------------- sinusoidal position add ----------------
__global__ void pos_add(float* __restrict__ x)
{
    long long id = (long long)blockIdx.x * 256 + threadIdx.x;
    if (id >= (long long)NTOK * 128) return;
    int c = (int)(id & 127); long long tok = id >> 7;
    int h = (int)((tok / WN) % HN); int w = (int)(tok % WN);
    int cc = c & 63;
    float e = (c < 64) ? ((h + 1) * TWO_PI / (HN + 1e-6f))
                       : ((w + 1) * TWO_PI / (WN + 1e-6f));
    float dim = powf(10000.f, (float)(2 * (cc >> 1)) / 64.f);
    float ang = e / dim;
    x[id] += (cc & 1) ? cosf(ang) : sinf(ang);
}

// ---------------- outlook: 9x9 sink softmax (in-place, ldc=128) ----------------
__global__ __launch_bounds__(256)
void outlook_sink(float* __restrict__ attn)
{
    int wv = threadIdx.x >> 5, lane = threadIdx.x & 31;
    long long pos = (long long)blockIdx.x * 8 + wv;
    __shared__ float a[8][81];
    float* A = attn + pos * 128;
    if (lane < 9) {           // column softmax over q (axis=-2), k = lane
        float col[9]; float m = -3.4e38f;
        #pragma unroll
        for (int q = 0; q < 9; ++q) { col[q] = A[q * 9 + lane]; m = fmaxf(m, col[q]); }
        float s = 0.f;
        #pragma unroll
        for (int q = 0; q < 9; ++q) { col[q] = expf(col[q] - m); s += col[q]; }
        float inv = 1.f / s;
        #pragma unroll
        for (int q = 0; q < 9; ++q) a[wv][q * 9 + lane] = col[q] * inv;
    }
    __syncthreads();
    if (lane < 9) {           // row normalize over k, q = lane
        float rs = 0.f;
        #pragma unroll
        for (int k = 0; k < 9; ++k) rs += a[wv][lane * 9 + k];
        float inv = 1.f / (rs + 1e-8f);
        #pragma unroll
        for (int k = 0; k < 9; ++k) A[lane * 9 + k] = a[wv][lane * 9 + k] * inv;
    }
}

// ---------------- outlook: attn(9x9) @ v_unf + overlapping fold (scatter) ----------------
__global__ __launch_bounds__(128)
void outlook_apply(const float* __restrict__ attn, const float* __restrict__ v,
                   float* __restrict__ y)
{
    int p = threadIdx.x >> 5, lane = threadIdx.x & 31;
    int pos = blockIdx.x * 4 + p;
    int b  = pos / (HN * WN);
    int h0 = (pos / WN) % HN;
    int w0 = pos % WN;
    __shared__ float at[4][81];
    for (int i = lane; i < 81; i += 32) at[p][i] = attn[(long long)pos * 128 + i];
    __syncthreads();
    int c4 = lane * 4;
    float4 vr[9];
    #pragma unroll
    for (int ki = 0; ki < 3; ++ki)
        #pragma unroll
        for (int kj = 0; kj < 3; ++kj) {
            int hh = h0 + ki - 1, ww = w0 + kj - 1;
            float4 val = {0.f, 0.f, 0.f, 0.f};
            if (hh >= 0 && hh < HN && ww >= 0 && ww < WN)
                val = *(const float4*)(v + (((long long)(b * HN + hh) * WN + ww) << 7) + c4);
            vr[ki * 3 + kj] = val;
        }
    #pragma unroll
    for (int i = 0; i < 3; ++i)
        #pragma unroll
        for (int j = 0; j < 3; ++j) {
            int q = i * 3 + j;
            float4 o = {0.f, 0.f, 0.f, 0.f};
            #pragma unroll
            for (int k = 0; k < 9; ++k) {
                float c = at[p][q * 9 + k];
                o.x += c * vr[k].x; o.y += c * vr[k].y; o.z += c * vr[k].z; o.w += c * vr[k].w;
            }
            int hd = h0 + i - 1, wd = w0 + j - 1;
            if (hd >= 0 && hd < HN && wd >= 0 && wd < WN) {
                float* yp = y + (((long long)(b * HN + hd) * WN + wd) << 7) + c4;
                atomicAdd(yp + 0, o.x); atomicAdd(yp + 1, o.y);
                atomicAdd(yp + 2, o.z); atomicAdd(yp + 3, o.w);
            }
        }
}

// ---------------- window pack (f32 linear -> bf16 windowed, optional roll) ----------------
__global__ void window_pack(const float* __restrict__ q, const float* __restrict__ k,
                            const float* __restrict__ v,
                            __bf16* __restrict__ qw, __bf16* __restrict__ kw,
                            __bf16* __restrict__ vw, int shift)
{
    long long id = (long long)blockIdx.x * 256 + threadIdx.x;
    if (id >= (long long)NTOK * 32) return;
    long long tok = id >> 5; int c4 = (int)(id & 31) * 4;
    int b = (int)(tok / (NS * NS * WTOK)); int r = (int)(tok % (NS * NS * WTOK));
    int win = r / WTOK; int n = r % WTOK;
    int wi = win / NS, wj = win % NS;
    int a = n / WW, bb = n % WW;
    int hs = wi * WH + a + (shift ? SH : 0); if (hs >= HN) hs -= HN;
    int xs = wj * WW + bb + (shift ? SW : 0); if (xs >= WN) xs -= WN;
    long long src = (((long long)(b * HN + hs) * WN + xs) << 7) + c4;
    long long dst = (tok << 7) + c4;
    float4 aq = *(const float4*)(q + src);
    float4 ak = *(const float4*)(k + src);
    float4 av = *(const float4*)(v + src);
    qw[dst + 0] = (__bf16)aq.x; qw[dst + 1] = (__bf16)aq.y; qw[dst + 2] = (__bf16)aq.z; qw[dst + 3] = (__bf16)aq.w;
    kw[dst + 0] = (__bf16)ak.x; kw[dst + 1] = (__bf16)ak.y; kw[dst + 2] = (__bf16)ak.z; kw[dst + 3] = (__bf16)ak.w;
    vw[dst + 0] = (__bf16)av.x; vw[dst + 1] = (__bf16)av.y; vw[dst + 2] = (__bf16)av.z; vw[dst + 3] = (__bf16)av.w;
}

// ---------------- window unpack (f32 windowed -> f32 linear, optional unroll) ----------------
__global__ void window_unpack(const float* __restrict__ aw, float* __restrict__ out, int shift)
{
    long long id = (long long)blockIdx.x * 256 + threadIdx.x;
    if (id >= (long long)NTOK * 32) return;
    long long tok = id >> 5; int c4 = (int)(id & 31) * 4;
    int b = (int)(tok / (HN * WN));
    int h = (int)((tok / WN) % HN);
    int w = (int)(tok % WN);
    int hc = h - (shift ? SH : 0); if (hc < 0) hc += HN;
    int wc = w - (shift ? SW : 0); if (wc < 0) wc += WN;
    int wi = hc / WH, a = hc % WH, wj = wc / WW, bb = wc % WW;
    long long src = (((long long)(b * NS * NS + wi * NS + wj) * WTOK + a * WW + bb) << 7) + c4;
    *(float4*)(out + (tok << 7) + c4) = *(const float4*)(aw + src);
}

// ---------------- sink softmax over windows: column stats (max, sum-exp over queries) ----------------
__global__ __launch_bounds__(256)
void sink_colstats(const float* __restrict__ S, float* __restrict__ cmax, float* __restrict__ csum)
{
    int c  = threadIdx.x & 63;
    int rg = threadIdx.x >> 6;
    int col = blockIdx.x * 64 + c;
    int win = blockIdx.y;
    const float* s = S + (long long)win * WTOK * WTOK;
    float m = -3.4e38f;
    for (int r = rg; r < WTOK; r += 4) m = fmaxf(m, s[(long long)r * WTOK + col]);
    __shared__ float red[4][64];
    red[rg][c] = m; __syncthreads();
    m = fmaxf(fmaxf(red[0][c], red[1][c]), fmaxf(red[2][c], red[3][c]));
    float sum = 0.f;
    for (int r = rg; r < WTOK; r += 4) sum += expf(s[(long long)r * WTOK + col] - m);
    __syncthreads();
    red[rg][c] = sum; __syncthreads();
    if (rg == 0) {
        float tot = red[0][c] + red[1][c] + red[2][c] + red[3][c];
        cmax[win * WTOK + col] = m;
        csum[win * WTOK + col] = tot;
    }
}

// ---------------- sink: a = exp(s-cmax)/csum, then row renormalize (in-place) ----------------
__global__ __launch_bounds__(256)
void sink_rownorm(float* __restrict__ S, const float* __restrict__ cmax,
                  const float* __restrict__ csum)
{
    int row = blockIdx.x, win = blockIdx.y;
    float* s = S + ((long long)win * WTOK + row) * WTOK;
    const float* cm = cmax + win * WTOK;
    const float* cs = csum + win * WTOK;
    float a[5]; int cnt = 0; float loc = 0.f;
    for (int m = threadIdx.x; m < WTOK; m += 256) {
        float v = expf(s[m] - cm[m]) / cs[m];
        a[cnt++] = v; loc += v;
    }
    __shared__ float red[256];
    red[threadIdx.x] = loc; __syncthreads();
    #pragma unroll
    for (int o = 128; o; o >>= 1) { if (threadIdx.x < o) red[threadIdx.x] += red[threadIdx.x + o]; __syncthreads(); }
    float inv = 1.f / (red[0] + 1e-8f);
    cnt = 0;
    for (int m = threadIdx.x; m < WTOK; m += 256) s[m] = a[cnt++] * inv;
}

// ======================= host orchestration =======================
extern "C" void kernel_launch(void* const* d_in, const int* in_sizes, int n_in,
                              void* d_out, int out_size, void* d_ws, size_t ws_size,
                              hipStream_t stream)
{
    auto F = [&](int i) { return (const float*)d_in[i]; };

    char* ws = (char*)d_ws;
    size_t off = 0;
    auto alloc = [&](size_t bytes) -> char* {
        char* p = ws + off; off = (off + bytes + 255) & ~(size_t)255; return p;
    };

    float*  X   = (float*) alloc((size_t)NTOK * 128 * 4);   // residual stream (NHWC)
    __bf16* XN  = (__bf16*)alloc((size_t)NTOK * 128 * 2);   // LN output
    float*  T0  = (float*) alloc((size_t)NTOK * 128 * 4);
    float*  T1  = (float*) alloc((size_t)NTOK * 128 * 4);
    float*  T2  = (float*) alloc((size_t)NTOK * 128 * 4);
    __bf16* QW  = (__bf16*)alloc((size_t)NTOK * 128 * 2);
    __bf16* KW  = (__bf16*)alloc((size_t)NTOK * 128 * 2);
    __bf16* VW  = (__bf16*)alloc((size_t)NTOK * 128 * 2);
    __bf16* HID = (__bf16*)alloc((size_t)NTOK * 512 * 2);
    float*  SC  = (float*) alloc((size_t)NWIN * WTOK * WTOK * 4);  // attention scores
    float*  CMX = (float*) alloc((size_t)NWIN * WTOK * 4);
    float*  CSM = (float*) alloc((size_t)NWIN * WTOK * 4);
    __bf16* WB  = (__bf16*)alloc((size_t)(11 * 16384 + 6 * 65536) * 2);

    __bf16* W_OWV = WB +  0 * 16384;
    __bf16* W_OWA = WB +  1 * 16384;   // 128 x (81 -> 128 padded)
    __bf16* W_OWO = WB +  2 * 16384;
    __bf16* W_WQ  = WB +  3 * 16384;
    __bf16* W_WK  = WB +  4 * 16384;
    __bf16* W_WV  = WB +  5 * 16384;
    __bf16* W_WO  = WB +  6 * 16384;
    __bf16* W_SQ  = WB +  7 * 16384;
    __bf16* W_SK  = WB +  8 * 16384;
    __bf16* W_SV  = WB +  9 * 16384;
    __bf16* W_SO  = WB + 10 * 16384;
    __bf16* W_F1A = WB + 11 * 16384;
    __bf16* W_F1B = W_F1A + 65536;
    __bf16* W_F2A = W_F1B + 65536;
    __bf16* W_F2B = W_F2A + 65536;
    __bf16* W_F3A = W_F2B + 65536;
    __bf16* W_F3B = W_F3A + 65536;

    auto cw = [&](int idx, __bf16* dst, int K, int N, int Npad) {
        long long tot = (long long)K * Npad;
        conv_w<<<dim3((unsigned)((tot + 255) / 256)), 256, 0, stream>>>(F(idx), dst, K, N, Npad);
    };
    // input order: left_feat, outlook{ln_g,ln_b,wv,bv,wa,ba,wo,bo}, ffn1{..}, win{..}, ffn2, swin, ffn3
    cw(3,  W_OWV, 128, 128, 128);
    cw(5,  W_OWA, 128,  81, 128);
    cw(7,  W_OWO, 128, 128, 128);
    cw(17, W_WQ,  128, 128, 128); cw(19, W_WK, 128, 128, 128);
    cw(21, W_WV,  128, 128, 128); cw(23, W_WO, 128, 128, 128);
    cw(33, W_SQ,  128, 128, 128); cw(35, W_SK, 128, 128, 128);
    cw(37, W_SV,  128, 128, 128); cw(39, W_SO, 128, 128, 128);
    cw(11, W_F1A, 128, 512, 512); cw(13, W_F1B, 512, 128, 128);
    cw(27, W_F2A, 128, 512, 512); cw(29, W_F2B, 512, 128, 128);
    cw(43, W_F3A, 128, 512, 512); cw(45, W_F3B, 512, 128, 128);

    nchw_to_nhwc<<<dim3(18432), 256, 0, stream>>>(F(0), X);

    auto gemm = [&](const void* A, const __bf16* B, const float* bias,
                    float* Cf, __bf16* Ch, const float* res,
                    int M, int N, int K, int lda, int ldb, int ldc,
                    long long sA, long long sB, long long sC, int batch,
                    float scale, int flags) {
        dim3 g((N + 127) / 128, M / 128, batch);
        const bool aF32 = (flags & GF_A_F32) != 0, bNT = (flags & GF_B_NT) != 0;
        if (aF32 && bNT)
            gemm_kernel<true, true><<<g, 256, 0, stream>>>(A, B, bias, Cf, Ch, res, M, N, K,
                                                           lda, ldb, ldc, sA, sB, sC, scale, flags);
        else if (aF32)
            gemm_kernel<true, false><<<g, 256, 0, stream>>>(A, B, bias, Cf, Ch, res, M, N, K,
                                                            lda, ldb, ldc, sA, sB, sC, scale, flags);
        else if (bNT)
            gemm_kernel<false, true><<<g, 256, 0, stream>>>(A, B, bias, Cf, Ch, res, M, N, K,
                                                            lda, ldb, ldc, sA, sB, sC, scale, flags);
        else
            gemm_kernel<false, false><<<g, 256, 0, stream>>>(A, B, bias, Cf, Ch, res, M, N, K,
                                                             lda, ldb, ldc, sA, sB, sC, scale, flags);
    };
    auto LN = [&](const float* g, const float* b) {
        ln_kernel<<<dim3(NTOK / 8), 256, 0, stream>>>(X, g, b, XN, NTOK);
    };
    auto FFN = [&](int i0, const __bf16* w1, const __bf16* w2) {
        LN(F(i0), F(i0 + 1));
        gemm(XN, w1, F(i0 + 3), nullptr, HID, nullptr, NTOK, 512, 128, 128, 512, 512,
             0, 0, 0, 1, 1.f, GF_GELU);
        gemm(HID, w2, F(i0 + 5), X, nullptr, X, NTOK, 128, 512, 512, 128, 128,
             0, 0, 0, 1, 1.f, GF_RES);
    };
    auto ATTN = [&](int i0, const __bf16* wq, const __bf16* wk, const __bf16* wv,
                    const __bf16* wo, int shift) {
        LN(F(i0), F(i0 + 1));
        gemm(XN, wq, F(i0 + 3), T0, nullptr, nullptr, NTOK, 128, 128, 128, 128, 128, 0, 0, 0, 1, 1.f, 0);
        gemm(XN, wk, F(i0 + 5), T1, nullptr, nullptr, NTOK, 128, 128, 128, 128, 128, 0, 0, 0, 1, 1.f, 0);
        gemm(XN, wv, F(i0 + 7), T2, nullptr, nullptr, NTOK, 128, 128, 128, 128, 128, 0, 0, 0, 1, 1.f, 0);
        window_pack<<<dim3(4608), 256, 0, stream>>>(T0, T1, T2, QW, KW, VW, shift);
        // scores = Qw @ Kw^T / sqrt(C) (+mask): batched NT GEMM (async-LDS pipeline if available)
        gemm(QW, KW, nullptr, SC, nullptr, nullptr, WTOK, WTOK, 128, 128, 128, WTOK,
             (long long)WTOK * 128, (long long)WTOK * 128, (long long)WTOK * WTOK, NWIN,
             0.08838834764831843f, GF_B_NT | (shift ? GF_MASK : 0));
        sink_colstats<<<dim3(WTOK / 64, NWIN), 256, 0, stream>>>(SC, CMX, CSM);
        sink_rownorm<<<dim3(WTOK, NWIN), 256, 0, stream>>>(SC, CMX, CSM);
        // out = attn @ Vw : A is f32 probs converted while staging LDS
        gemm(SC, VW, nullptr, T0, nullptr, nullptr, WTOK, 128, WTOK, WTOK, 128, 128,
             (long long)WTOK * WTOK, (long long)WTOK * 128, (long long)WTOK * 128, NWIN,
             1.f, GF_A_F32);
        window_unpack<<<dim3(4608), 256, 0, stream>>>(T0, T1, shift);
        gemm(T1, wo, F(i0 + 9), X, nullptr, X, NTOK, 128, 128, 128, 128, 128,
             0, 0, 0, 1, 1.f, GF_A_F32 | GF_RES);
    };

    // ---- outlook attention ----
    LN(F(1), F(2));
    gemm(XN, W_OWV, F(4), T0, nullptr, nullptr, NTOK, 128, 128, 128, 128, 128, 0, 0, 0, 1, 1.f, 0);
    gemm(XN, W_OWA, F(6), T1, nullptr, nullptr, NTOK,  81, 128, 128, 128, 128, 0, 0, 0, 1, 1.f, 0);
    outlook_sink<<<dim3(NTOK / 8), 256, 0, stream>>>(T1);
    fill_zero<<<dim3(18432), 256, 0, stream>>>(T2, (long long)NTOK * 128);
    outlook_apply<<<dim3(NTOK / 4), 128, 0, stream>>>(T1, T0, T2);
    gemm(T2, W_OWO, F(8), X, nullptr, X, NTOK, 128, 128, 128, 128, 128, 0, 0, 0, 1, 1.f,
         GF_A_F32 | GF_RES);

    FFN(9, W_F1A, W_F1B);
    pos_add<<<dim3(18432), 256, 0, stream>>>(X);
    ATTN(15, W_WQ, W_WK, W_WV, W_WO, 0);
    FFN(25, W_F2A, W_F2B);
    ATTN(31, W_SQ, W_SK, W_SV, W_SO, 1);
    FFN(41, W_F3A, W_F3B);

    nhwc_to_nchw<<<dim3(18432), 256, 0, stream>>>(X, (float*)d_out);

    (void)in_sizes; (void)n_in; (void)out_size; (void)ws_size;
}
